// VolumetricLogic_8761733283919
// MI455X (gfx1250) — matrix operand order repaired and checked by reading, not compiled, not run on hardware
//
#include <hip/hip_runtime.h>

// ---------- problem dimensions (fixed by the reference) ----------
#define N_DIM 16384   // B*S
#define D_DIM 1024
#define K_DIM 4096

typedef __attribute__((ext_vector_type(16))) __bf16 bf16x16;
typedef __attribute__((ext_vector_type(8)))  __bf16 bf16x8;
typedef __attribute__((ext_vector_type(4)))  __bf16 bf16x4;
typedef __attribute__((ext_vector_type(8)))  float  v8f;

// =================================================================
// Kernel 1: row-wise L2 normalize (f32 -> bf16).  One wave per row,
// cols fixed at 1024: each lane handles 32 elements (8x float4).
// =================================================================
__global__ void l2norm_rows_bf16(const float* __restrict__ src,
                                 __bf16* __restrict__ dst) {
    const int cols = D_DIM;
    int row  = blockIdx.x * 8 + (threadIdx.x >> 5);   // 8 waves / block
    int lane = threadIdx.x & 31;

    const float4* s = (const float4*)(src + (size_t)row * cols);
    float4 v[8];
    float ssq = 0.f;
#pragma unroll
    for (int i = 0; i < 8; ++i) {
        v[i] = s[lane + i * 32];
        ssq += v[i].x * v[i].x + v[i].y * v[i].y +
               v[i].z * v[i].z + v[i].w * v[i].w;
    }
#pragma unroll
    for (int off = 16; off > 0; off >>= 1)
        ssq += __shfl_xor(ssq, off, 32);

    float scale = 1.0f / fmaxf(sqrtf(ssq), 1e-12f);

    bf16x4* d = (bf16x4*)(dst + (size_t)row * cols);
#pragma unroll
    for (int i = 0; i < 8; ++i) {
        bf16x4 o;
        o[0] = (__bf16)(v[i].x * scale);
        o[1] = (__bf16)(v[i].y * scale);
        o[2] = (__bf16)(v[i].z * scale);
        o[3] = (__bf16)(v[i].w * scale);
        d[lane + i * 32] = o;
    }
}

// =================================================================
// Kernel 2: inv_r[k] = 1 / (softplus(log_radii[k]) + 0.01)
// =================================================================
__global__ void radii_kernel(const float* __restrict__ log_radii,
                             float* __restrict__ inv_r) {
    int k = blockIdx.x * blockDim.x + threadIdx.x;
    if (k < K_DIM) {
        float x  = log_radii[k];
        float sp = fmaxf(x, 0.f) + log1pf(expf(-fabsf(x)));  // stable softplus
        inv_r[k] = 1.0f / (sp + 0.01f);
    }
}

// =================================================================
// Kernel 3: p_t[d*K + k] = bf16( push[k*D + d] * out_scale[d] )
// LDS 32x32 tile transpose; folds out_scale into the B operand of
// the second GEMM so the final store has no extra pass.
// =================================================================
__global__ void transpose_scale_push(const float* __restrict__ push,
                                     const float* __restrict__ out_scale,
                                     __bf16* __restrict__ p_t) {
    __shared__ float tile[32][33];
    int kb = blockIdx.x * 32;   // along K
    int db = blockIdx.y * 32;   // along D
    // load push[kb+i][db+x]  (coalesced along d)
    for (int i = threadIdx.y; i < 32; i += 8)
        tile[i][threadIdx.x] = push[(size_t)(kb + i) * D_DIM + db + threadIdx.x];
    __syncthreads();
    // write p_t[db+i][kb+x]  (coalesced along k)
    for (int i = threadIdx.y; i < 32; i += 8) {
        float s = out_scale[db + i];
        p_t[(size_t)(db + i) * K_DIM + kb + threadIdx.x] =
            (__bf16)(tile[threadIdx.x][i] * s);
    }
}

// =================================================================
// WMMA helpers: fragment loads; A and B share contiguous k order (k = hi*16 + s).
//  A (16x32 bf16): lane r=lane&15 -> row M=r; hi=lane>>4.
//    elems 0..7  = K: base + hi*16 .. +7
//    elems 8..15 = K: base + hi*16 + 8 .. +7
//  B (32x16 bf16): lane r -> column N=r; elems 0..15 = K: base+hi*16..
//  C/D (16x16 f32): n = lane&15 ; m = vgpr + (lane>=16 ? 8 : 0)
// =================================================================
__device__ __forceinline__ bf16x16 load_frag_A(const __bf16* base) {
    bf16x8 lo = *(const bf16x8*)(base);
    bf16x8 hi = *(const bf16x8*)(base + 8);
    return __builtin_shufflevector(lo, hi, 0, 1, 2, 3, 4, 5, 6, 7,
                                           8, 9, 10, 11, 12, 13, 14, 15);
}
__device__ __forceinline__ bf16x16 load_frag_B(const __bf16* base) {
    return *(const bf16x16*)(base);
}

// =================================================================
// Kernel 4: phi[N,K] (bf16) = relu(1 + (h_nb @ c_nb^T - 1) * inv_r)
// Block = 8 waves arranged as a 2(M) x 4(N) tile grid covering a
// 128x256 output macro-tile; each wave computes 64x64 (16 WMMA accum
// tiles).  A rows are shared by 4 waves, B rows by 2 waves -> both
// operands get cross-wave reuse through the WGP$ without LDS traffic.
// Contraction over D in steps of 32 with v_wmma_f32_16x16x32_bf16.
// =================================================================
__global__ void __launch_bounds__(256)
gemm_phi(const __bf16* __restrict__ hN, const __bf16* __restrict__ cN,
         const float* __restrict__ inv_r, __bf16* __restrict__ phi) {
    const int KB = K_DIM / 256;                 // 16 block-cols
    int bm = blockIdx.x / KB;                   // macro-tile row  (128 rows)
    int bk = blockIdx.x % KB;                   // macro-tile col  (256 centers)
    int w  = threadIdx.x >> 5;
    int mbase = (bm * 2 + (w >> 2)) * 64;       // 2 wave-rows
    int kbase = (bk * 4 + (w & 3)) * 64;        // 4 wave-cols
    int lane  = threadIdx.x & 31;
    int r     = lane & 15;
    int hi    = lane >> 4;

    v8f acc[4][4] = {};

    const __bf16* aBase = hN + (size_t)(mbase + r) * D_DIM + hi * 16;
    const __bf16* bBase = cN + (size_t)(kbase + r) * D_DIM + hi * 16;

    for (int d = 0; d < D_DIM; d += 32) {
        bf16x16 a[4], b[4];
#pragma unroll
        for (int i = 0; i < 4; ++i) {
            a[i] = load_frag_A(aBase + (size_t)i * 16 * D_DIM + d);
            b[i] = load_frag_B(bBase + (size_t)i * 16 * D_DIM + d);
        }
#pragma unroll
        for (int i = 0; i < 4; ++i)
#pragma unroll
            for (int j = 0; j < 4; ++j)
                acc[i][j] = __builtin_amdgcn_wmma_f32_16x16x32_bf16(
                    false, a[i], false, b[j], (short)0, acc[i][j], false, false);
    }

    // epilogue: phi = max(1 + (sim-1)*inv_r, 0), store bf16
#pragma unroll
    for (int j = 0; j < 4; ++j) {
        int center = kbase + j * 16 + r;
        float ir = inv_r[center];
#pragma unroll
        for (int i = 0; i < 4; ++i) {
#pragma unroll
            for (int e = 0; e < 8; ++e) {
                int m = mbase + i * 16 + hi * 8 + e;
                float p = fmaxf(fmaf(acc[i][j][e] - 1.0f, ir, 1.0f), 0.0f);
                phi[(size_t)m * K_DIM + center] = (__bf16)p;
            }
        }
    }
}

// =================================================================
// Kernel 5: out[N,D] (f32) = phi[N,K] @ p_t^T   (p_t is [D,K], so this
// is again an NT GEMM with contiguous fragment loads). out_scale is
// already folded into p_t.  Same 2x4 wave grid per block for WGP$
// cross-wave operand reuse.
// =================================================================
__global__ void __launch_bounds__(256)
gemm_out(const __bf16* __restrict__ phi, const __bf16* __restrict__ p_t,
         float* __restrict__ out) {
    const int DB = D_DIM / 256;                 // 4 block-cols
    int bm = blockIdx.x / DB;                   // macro-tile row (128 rows)
    int bd = blockIdx.x % DB;                   // macro-tile col (256 d-cols)
    int w  = threadIdx.x >> 5;
    int mbase = (bm * 2 + (w >> 2)) * 64;
    int dbase = (bd * 4 + (w & 3)) * 64;
    int lane  = threadIdx.x & 31;
    int r     = lane & 15;
    int hi    = lane >> 4;

    v8f acc[4][4] = {};

    const __bf16* aBase = phi + (size_t)(mbase + r) * K_DIM + hi * 16;
    const __bf16* bBase = p_t + (size_t)(dbase + r) * K_DIM + hi * 16;

    for (int k = 0; k < K_DIM; k += 32) {
        bf16x16 a[4], b[4];
#pragma unroll
        for (int i = 0; i < 4; ++i) {
            a[i] = load_frag_A(aBase + (size_t)i * 16 * K_DIM + k);
            b[i] = load_frag_B(bBase + (size_t)i * 16 * K_DIM + k);
        }
#pragma unroll
        for (int i = 0; i < 4; ++i)
#pragma unroll
            for (int j = 0; j < 4; ++j)
                acc[i][j] = __builtin_amdgcn_wmma_f32_16x16x32_bf16(
                    false, a[i], false, b[j], (short)0, acc[i][j], false, false);
    }

#pragma unroll
    for (int j = 0; j < 4; ++j) {
        int dcol = dbase + j * 16 + r;
#pragma unroll
        for (int i = 0; i < 4; ++i) {
#pragma unroll
            for (int e = 0; e < 8; ++e) {
                int m = mbase + i * 16 + hi * 8 + e;
                out[(size_t)m * D_DIM + dcol] = acc[i][j][e];
            }
        }
    }
}

// =================================================================
// Host-side launcher
// =================================================================
extern "C" void kernel_launch(void* const* d_in, const int* in_sizes, int n_in,
                              void* d_out, int out_size, void* d_ws, size_t ws_size,
                              hipStream_t stream) {
    const float* h         = (const float*)d_in[0];   // [4,4096,1024]
    const float* centers   = (const float*)d_in[1];   // [4096,1024]
    const float* log_radii = (const float*)d_in[2];   // [4096]
    const float* push      = (const float*)d_in[3];   // [4096,1024]
    const float* out_scale = (const float*)d_in[4];   // [1024]
    float* out             = (float*)d_out;           // [4,4096,1024] f32

    // ---- workspace carve-out (256B aligned) ----
    char*  ws  = (char*)d_ws;
    size_t off = 0;
    auto carve = [&](size_t bytes) -> void* {
        void* p = ws + off;
        off += (bytes + 255) & ~(size_t)255;
        return p;
    };
    __bf16* h_nb  = (__bf16*)carve((size_t)N_DIM * D_DIM * 2);  // 32 MB
    __bf16* c_nb  = (__bf16*)carve((size_t)K_DIM * D_DIM * 2);  //  8 MB
    __bf16* p_t   = (__bf16*)carve((size_t)D_DIM * K_DIM * 2);  //  8 MB
    float*  inv_r = (float*) carve((size_t)K_DIM * 4);
    __bf16* phi   = (__bf16*)carve((size_t)N_DIM * K_DIM * 2);  // 128 MB
    (void)ws_size;

    // 1) normalize h and centers -> bf16
    hipLaunchKernelGGL(l2norm_rows_bf16, dim3(N_DIM / 8), dim3(256), 0, stream,
                       h, h_nb);
    hipLaunchKernelGGL(l2norm_rows_bf16, dim3(K_DIM / 8), dim3(256), 0, stream,
                       centers, c_nb);

    // 2) inverse radii
    hipLaunchKernelGGL(radii_kernel, dim3((K_DIM + 255) / 256), dim3(256), 0,
                       stream, log_radii, inv_r);

    // 3) push^T * out_scale -> bf16 [D,K]
    hipLaunchKernelGGL(transpose_scale_push, dim3(K_DIM / 32, D_DIM / 32),
                       dim3(32, 8), 0, stream, push, out_scale, p_t);

    // 4) GEMM1 + phi epilogue : (N/128)*(K/256) = 2048 blocks of 8 waves
    hipLaunchKernelGGL(gemm_phi, dim3((N_DIM / 128) * (K_DIM / 256)),
                       dim3(256), 0, stream, h_nb, c_nb, inv_r, phi);

    // 5) GEMM2 : (N/128)*(D/256) = 512 blocks of 8 waves
    hipLaunchKernelGGL(gemm_out, dim3((N_DIM / 128) * (D_DIM / 256)),
                       dim3(256), 0, stream, phi, p_t, out);
}
